// MambaConditioningBlock_81140522156518
// MI455X (gfx1250) — matrix operand order repaired and checked by reading, not compiled, not run on hardware
//
#include <hip/hip_runtime.h>
#include <hip/hip_bf16.h>
#include <math.h>

// Problem dims (fixed by reference)
#define BB      4
#define LL      2048
#define DM      1024          // d_model
#define DI      2048          // d_inner
#define DTR     64            // dt_rank
#define DSTATE  16
#define MROWS   (BB * LL)     // 8192

#define TILE_K  32
#define TILE_N  32

typedef __attribute__((ext_vector_type(16))) _Float16 v16h;
typedef __attribute__((ext_vector_type(8)))  float    v8f;

__device__ __forceinline__ float sigmoidf_(float x) { return 1.0f / (1.0f + __expf(-x)); }
__device__ __forceinline__ float siluf_(float x)    { return x * sigmoidf_(x); }

// ---------------------------------------------------------------------------
// 1) sc = silu(c)   [4, 1024]
// ---------------------------------------------------------------------------
__global__ void k_silu_c(const float* __restrict__ c, float* __restrict__ sc) {
    int i = blockIdx.x * 256 + threadIdx.x;     // < 4096
    sc[i] = siluf_(c[i]);
}

// ---------------------------------------------------------------------------
// 2) mod = sc @ W_ada + b_ada   [4, 3072]
// ---------------------------------------------------------------------------
__global__ void k_mod(const float* __restrict__ sc, const float* __restrict__ W_ada,
                      const float* __restrict__ b_ada, float* __restrict__ mod) {
    int idx = blockIdx.x * 256 + threadIdx.x;   // < 12288
    int b = idx / 3072, j = idx % 3072;
    float acc = b_ada[j];
    const float* srow = sc + b * DM;
    for (int k = 0; k < DM; ++k)
        acc = fmaf(srow[k], W_ada[(size_t)k * 3072 + j], acc);
    mod[idx] = acc;
}

// ---------------------------------------------------------------------------
// 3) LayerNorm (no affine) + adaLN modulate: h = xn*(1+scale) + shift
// ---------------------------------------------------------------------------
__global__ __launch_bounds__(256)
void k_ln_mod(const float* __restrict__ x, const float* __restrict__ mod,
              float* __restrict__ h) {
    __shared__ float s1[256], s2[256];
    const int row = blockIdx.x;           // b*L + l
    const int b   = row >> 11;            // / 2048
    const int tid = threadIdx.x;
    const float* xr = x + (size_t)row * DM;
    float v[4], sum = 0.f, sq = 0.f;
#pragma unroll
    for (int i = 0; i < 4; ++i) {
        v[i] = xr[tid + 256 * i];
        sum += v[i];
        sq  = fmaf(v[i], v[i], sq);
    }
    s1[tid] = sum; s2[tid] = sq;
    __syncthreads();
    for (int o = 128; o > 0; o >>= 1) {
        if (tid < o) { s1[tid] += s1[tid + o]; s2[tid] += s2[tid + o]; }
        __syncthreads();
    }
    const float mean = s1[0] * (1.0f / DM);
    const float var  = s2[0] * (1.0f / DM) - mean * mean;
    const float rstd = rsqrtf(var + 1e-6f);
    float* hr = h + (size_t)row * DM;
#pragma unroll
    for (int i = 0; i < 4; ++i) {
        int col = tid + 256 * i;
        float xn    = (v[i] - mean) * rstd;
        float scale = mod[b * 3072 + col];
        float shift = mod[b * 3072 + DM + col];
        hr[col] = fmaf(xn, 1.0f + scale, shift);
    }
}

// ---------------------------------------------------------------------------
// 4) WMMA GEMM: C[M,N] = A[M,K] @ W[K,N]  (f32 in/out, f16 WMMA, f32 accum)
//    Block = 256 threads = 8 wave32s; wave w -> 16x32 output tile.
//    Block tile = 128(M) x 32(N); B tile staged once per block into LDS via
//    global_load_async_to_lds_b128 (ASYNCcnt path) on gfx1250.
//    Requires M%128==0, N%32==0, K%32==0.
// ---------------------------------------------------------------------------
__global__ __launch_bounds__(256)
void k_gemm(const float* __restrict__ A, const float* __restrict__ W,
            float* __restrict__ C, int K, int N, int lda, int ldc) {
    __shared__ __align__(16) float bsm[TILE_K * TILE_N];   // 4 KB B tile

    const int tid  = threadIdx.x;
    const int lane = tid & 31;
    const int wave = tid >> 5;
    const int m0   = blockIdx.x * 128 + wave * 16;
    const int n0   = blockIdx.y * TILE_N;

    const int mrow = m0 + (lane & 15);
    const int kh_a = (lane >> 4) << 4;    // A half: lanes 0-15 -> K 0..15; 16-31 -> K 16..31
    const int kh_b = (lane >> 4) << 4;    // B half: lanes 0-15 -> K 0..15; 16-31 -> K 16..31
    const int nbl  = lane & 15;           // B fragment column within tile

    // B-tile staging assignment: one 16-byte chunk per thread (256 * 16B = 4 KB)
    const int crow = tid >> 3;            // 0..31   (k row within tile)
    const int ccol = (tid & 7) << 2;      // 0,4,..,28 (float column within tile)

    const float* Ap = A + (size_t)mrow * lda + kh_a;
    v8f acc0 = {}; v8f acc1 = {};

    for (int k0 = 0; k0 < K; k0 += TILE_K) {
        // ---- stage B tile (32K x 32N f32) into LDS, shared by all 8 waves ----
        const float* gsrc = W + (size_t)(k0 + crow) * N + (n0 + ccol);
        float*       ldst = &bsm[crow * TILE_N + ccol];
#if defined(__gfx1250__)
        {
            // Low 32 bits of the generic (flat) address of a __shared__ object
            // ARE the wave-relative LDS byte offset.
            const unsigned int       lds_off = (unsigned int)(uintptr_t)ldst;
            const unsigned long long gaddr   = (unsigned long long)(uintptr_t)gsrc;
            asm volatile("global_load_async_to_lds_b128 %0, %1, off"
                         :: "v"(lds_off), "v"(gaddr)
                         : "memory");
            asm volatile("s_wait_asynccnt 0x0" ::: "memory");
        }
#else
        *(float4*)ldst = *(const float4*)gsrc;
#endif
        __syncthreads();

        // ---- A fragment: 16 f16 per lane (two contiguous 8-float runs) ----
        v16h a;
        {
            const float* p = Ap + k0;
            __builtin_prefetch(p + TILE_K, 0, 0);     // global_prefetch_b8 (next K tile)
#pragma unroll
            for (int j = 0; j < 8; ++j) a[j]     = (_Float16)p[j];
#pragma unroll
            for (int j = 0; j < 8; ++j) a[8 + j] = (_Float16)p[8 + j];
        }
        // ---- B fragments from LDS: column nbl / nbl+16, K = kh_b + j ----
        v16h b0, b1;
#pragma unroll
        for (int j = 0; j < 16; ++j) {
            const float* r = &bsm[(kh_b + j) * TILE_N];
            b0[j] = (_Float16)r[nbl];
            b1[j] = (_Float16)r[nbl + 16];
        }
        acc0 = __builtin_amdgcn_wmma_f32_16x16x32_f16(false, a, false, b0,
                                                      (short)0, acc0, false, false);
        acc1 = __builtin_amdgcn_wmma_f32_16x16x32_f16(false, a, false, b1,
                                                      (short)0, acc1, false, false);
        __syncthreads();
    }

    // ---- C store: VGPR i -> row m0 + i + 8*(lane>=16), col = lane&15 ----
    const int rbase = m0 + ((lane >> 4) << 3);
    const int nb    = n0 + nbl;
#pragma unroll
    for (int i = 0; i < 8; ++i) {
        C[(size_t)(rbase + i) * ldc + nb]      = acc0[i];
        C[(size_t)(rbase + i) * ldc + nb + 16] = acc1[i];
    }
}

// ---------------------------------------------------------------------------
// 5) causal depthwise conv (kernel 4) on xm (= xz cols 0..2047) + bias + silu
// ---------------------------------------------------------------------------
__global__ void k_conv_silu(const float* __restrict__ xz, const float* __restrict__ conv_w,
                            const float* __restrict__ conv_b, float* __restrict__ xc) {
    int idx = blockIdx.x * 256 + threadIdx.x;   // < 8192*2048
    int d = idx & (DI - 1);
    int r = idx >> 11;                          // b*L + l
    int l = r & (LL - 1);
    float acc = conv_b[d];
#pragma unroll
    for (int k = 0; k < 4; ++k) {
        int ls = l + k - 3;
        if (ls >= 0)
            acc = fmaf(xz[(size_t)(r + k - 3) * (2 * DI) + d], conv_w[d * 4 + k], acc);
    }
    xc[(size_t)r * DI + d] = siluf_(acc);
}

// ---------------------------------------------------------------------------
// 6) delta = softplus(draw + b_dt)  (in place)
// ---------------------------------------------------------------------------
__global__ void k_delta(float* __restrict__ draw, const float* __restrict__ b_dt) {
    int idx = blockIdx.x * 256 + threadIdx.x;   // < 8192*2048
    int d = idx & (DI - 1);
    float v = draw[idx] + b_dt[d];
    draw[idx] = (v > 20.f) ? v : log1pf(__expf(v));
}

// ---------------------------------------------------------------------------
// 7) selective scan; one lane per (b, channel). State (16 floats) in VGPRs.
//    Fuses +u*D_skip and *silu(z); writes y2 in place over xc.
// ---------------------------------------------------------------------------
__global__ __launch_bounds__(256)
void k_scan(const float* __restrict__ delta, float* __restrict__ xc,
            const float* __restrict__ xz,   const float* __restrict__ dbc,
            const float* __restrict__ A_log, const float* __restrict__ D_skip) {
    const int d = blockIdx.x * 256 + threadIdx.x;   // channel 0..2047
    const int b = blockIdx.y;                       // batch
    float Arow[DSTATE];
#pragma unroll
    for (int s = 0; s < DSTATE; ++s) Arow[s] = -__expf(A_log[d * DSTATE + s]);
    const float Dd = D_skip[d];
    float h[DSTATE];
#pragma unroll
    for (int s = 0; s < DSTATE; ++s) h[s] = 0.f;

    for (int l = 0; l < LL; ++l) {
        const size_t r = (size_t)b * LL + l;
        const float dt = delta[r * DI + d];
        const float u  = xc[r * DI + d];
        const float z  = xz[r * (2 * DI) + DI + d];
        const float* bc = dbc + r * 96;             // B at +64, C at +80 (wave-uniform)
        const float du = dt * u;
        float y = 0.f;
#pragma unroll
        for (int s = 0; s < DSTATE; ++s) {
            float dA = __expf(dt * Arow[s]);
            h[s] = fmaf(dA, h[s], du * bc[DTR + s]);
            y = fmaf(h[s], bc[DTR + DSTATE + s], y);
        }
        y = fmaf(u, Dd, y);
        xc[r * DI + d] = y * siluf_(z);             // y2 in place
    }
}

// ---------------------------------------------------------------------------
// 8) out = x + gate * out   (in place on d_out)
// ---------------------------------------------------------------------------
__global__ void k_final(const float* __restrict__ x, const float* __restrict__ mod,
                        float* __restrict__ out) {
    int idx = blockIdx.x * 256 + threadIdx.x;   // < 8192*1024
    int j = idx & (DM - 1);
    int b = idx >> 21;                          // / (L*DM)
    out[idx] = fmaf(mod[b * 3072 + 2 * DM + j], out[idx], x[idx]);
}

// ---------------------------------------------------------------------------
extern "C" void kernel_launch(void* const* d_in, const int* in_sizes, int n_in,
                              void* d_out, int out_size, void* d_ws, size_t ws_size,
                              hipStream_t stream) {
    const float* x      = (const float*)d_in[0];
    const float* c      = (const float*)d_in[1];
    const float* W_in   = (const float*)d_in[2];
    const float* conv_w = (const float*)d_in[3];
    const float* conv_b = (const float*)d_in[4];
    const float* W_xproj= (const float*)d_in[5];
    const float* W_dt   = (const float*)d_in[6];
    const float* b_dt   = (const float*)d_in[7];
    const float* A_log  = (const float*)d_in[8];
    const float* D_skip = (const float*)d_in[9];
    const float* W_out  = (const float*)d_in[10];
    const float* W_ada  = (const float*)d_in[11];
    const float* b_ada  = (const float*)d_in[12];
    float* out = (float*)d_out;

    // Workspace layout (floats): ~76.3M floats (~292 MB)
    float* w    = (float*)d_ws;
    float* sc   = w;                                  //  4 * 1024
    float* mod  = sc   + (size_t)BB * DM;             //  4 * 3072
    float* hbuf = mod  + (size_t)BB * 3 * DM;         //  8192 * 1024
    float* xz   = hbuf + (size_t)MROWS * DM;          //  8192 * 4096
    float* xc   = xz   + (size_t)MROWS * 2 * DI;      //  8192 * 2048
    float* dbc  = xc   + (size_t)MROWS * DI;          //  8192 * 96
    float* draw = dbc  + (size_t)MROWS * 96;          //  8192 * 2048

    // conditioning
    k_silu_c<<<(BB * DM) / 256, 256, 0, stream>>>(c, sc);
    k_mod   <<<(BB * 3 * DM) / 256, 256, 0, stream>>>(sc, W_ada, b_ada, mod);
    k_ln_mod<<<MROWS, 256, 0, stream>>>(x, mod, hbuf);

    // in-projection: xz = h @ W_in   [8192,1024]x[1024,4096]
    k_gemm<<<dim3(MROWS / 128, (2 * DI) / 32), 256, 0, stream>>>(
        hbuf, W_in, xz, DM, 2 * DI, DM, 2 * DI);

    // depthwise causal conv + silu
    k_conv_silu<<<(MROWS * DI) / 256, 256, 0, stream>>>(xz, conv_w, conv_b, xc);

    // x-projection: dbc = xc @ W_xproj   [8192,2048]x[2048,96]
    k_gemm<<<dim3(MROWS / 128, 96 / 32), 256, 0, stream>>>(
        xc, W_xproj, dbc, DI, 96, DI, 96);

    // dt projection: draw = dbc[:, :64] @ W_dt   [8192,64]x[64,2048]
    k_gemm<<<dim3(MROWS / 128, DI / 32), 256, 0, stream>>>(
        dbc, W_dt, draw, DTR, DI, 96, DI);

    // delta = softplus(draw + b_dt)
    k_delta<<<(MROWS * DI) / 256, 256, 0, stream>>>(draw, b_dt);

    // selective scan (fused skip + gating), y2 overwrites xc
    k_scan<<<dim3(DI / 256, BB), 256, 0, stream>>>(draw, xc, xz, dbc, A_log, D_skip);

    // out-projection straight into d_out: out = y2 @ W_out  [8192,2048]x[2048,1024]
    k_gemm<<<dim3(MROWS / 128, DM / 32), 256, 0, stream>>>(
        xc, W_out, out, DI, DM, DI, DM);

    // residual + gate (in place)
    k_final<<<(MROWS * DM) / 256, 256, 0, stream>>>(x, mod, out);
}
